// GraphAttentionV2Layer_47605417509431
// MI455X (gfx1250) — matrix operand order repaired and checked by reading, not compiled, not run on hardware
//
#include <hip/hip_runtime.h>
#include <hip/hip_bf16.h>

typedef _Float16 v16h __attribute__((ext_vector_type(16)));
typedef _Float16 v8h  __attribute__((ext_vector_type(8)));
typedef float    v8f  __attribute__((ext_vector_type(8)));
typedef unsigned int tdm_u4 __attribute__((ext_vector_type(4)));
typedef int          tdm_i8 __attribute__((ext_vector_type(8)));
typedef int          tdm_i4 __attribute__((ext_vector_type(4)));

union Frag { v16h v; v8h h[2]; };

#define NB   400          // B*SL
#define NN   128          // nodes
#define NH   8            // heads
#define FH_  32           // head dim
#define QS_STRIDE 264     // 256 + 8 pad (halves)
#define AS_STRIDE 136     // 128 + 8 pad (halves)
#define SS_STRIDE 40      // 32 + 8 pad (halves)
#define SMEM_BYTES 276480
#define INV_SCALE 0.005524271728019903f   // 1/sqrt(32*8*128)

// ---------------- ws half-index layout ----------------
// [0)       Wl f16  [256][128]   32768 halves
// [32768)   -Wr f16 [256][128]
// [65536)   Wv f16  [256][128]
// [98304)   attS f16 [32][128]   (att_w[:, :128])
// [102400)  attV f16 [32][32]    (att_w[:, 128:])
// [103424)  ff_w1 f16 [32][32]
// [104448)  ff_w2 f16 [32][32]
// byte 210944: float region: P[128], R[128], LNB[128], Sww, Swb, Sbb

__global__ __launch_bounds__(256) void gat_prep(
    const float* __restrict__ Wl, const float* __restrict__ Wr, const float* __restrict__ Wv,
    const float* __restrict__ sa_w, const float* __restrict__ sa_b,
    const float* __restrict__ ln_w, const float* __restrict__ ln_b,
    const float* __restrict__ att_w, const float* __restrict__ ff_w1,
    const float* __restrict__ ff_w2, char* __restrict__ ws)
{
    _Float16* f = (_Float16*)ws;
    int t = threadIdx.x;
    for (int idx = t; idx < 32768; idx += 256) {
        f[idx]         = (_Float16)Wl[idx];
        f[32768 + idx] = (_Float16)(-Wr[idx]);   // k = (-h0) @ Wr^T
        f[65536 + idx] = (_Float16)Wv[idx];
    }
    for (int idx = t; idx < 32 * 160; idx += 256) {
        int row = idx / 160, col = idx % 160;
        float x = att_w[idx];
        if (col < 128) f[98304 + row * 128 + col] = (_Float16)x;
        else           f[102400 + row * 32 + (col - 128)] = (_Float16)x;
    }
    for (int idx = t; idx < 1024; idx += 256) {
        f[103424 + idx] = (_Float16)ff_w1[idx];
        f[104448 + idx] = (_Float16)ff_w2[idx];
    }
    float* g = (float*)(ws + 210944);
    __shared__ float mwb[2];
    if (t == 0) {
        float mw = 0.f, mb = 0.f;
        for (int j = 0; j < 128; j++) { mw += sa_w[j]; mb += sa_b[j]; }
        mw *= (1.f / 128.f); mb *= (1.f / 128.f);
        mwb[0] = mw; mwb[1] = mb;
        float Sww = 0.f, Swb = 0.f, Sbb = 0.f;
        for (int j = 0; j < 128; j++) {
            float dw = sa_w[j] - mw, db = sa_b[j] - mb;
            Sww += dw * dw; Swb += dw * db; Sbb += db * db;
        }
        g[384] = Sww * (1.f / 128.f);
        g[385] = Swb * (1.f / 128.f);
        g[386] = Sbb * (1.f / 128.f);
    }
    __syncthreads();
    if (t < 128) {
        float mw = mwb[0], mb = mwb[1];
        g[t]       = (sa_w[t] - mw) * ln_w[t];   // P_j
        g[128 + t] = (sa_b[t] - mb) * ln_w[t];   // R_j
        g[256 + t] = ln_b[t];                    // LNB_j
    }
}

__device__ __forceinline__ v8f wmma_f16(const Frag& a, const Frag& b, v8f c) {
    return __builtin_amdgcn_wmma_f32_16x16x32_f16(false, a.v, false, b.v,
                                                  (short)0, c, false, false);
}

#if defined(__has_builtin)
#if __has_builtin(__builtin_amdgcn_tensor_load_to_lds)
#define HAVE_TDM 1
#endif
#endif

#ifdef HAVE_TDM
// Issue a 2D 128x128 f32 TDM copy global -> LDS (D# per CDNA5 ISA sec 8.3/8.4).
__device__ __forceinline__ void tdm_load_128x128_f32(const void* gptr, unsigned lds_byte_off) {
    unsigned long long ga = (unsigned long long)gptr;
    tdm_u4 g0;
    g0[0] = 1u;                                   // count=1, user descriptor
    g0[1] = lds_byte_off;                         // lds_addr
    g0[2] = (unsigned)(ga & 0xffffffffu);         // global_addr[31:0]
    g0[3] = (unsigned)((ga >> 32) & 0x1ffffffu)   // global_addr[56:32]
          | (2u << 30);                           // type=2 (image)
    tdm_i8 g1;
    g1[0] = 0x00020000;       // workgroup_mask=0, data_size=2 (4B)
    g1[1] = 128 << 16;        // tensor_dim0[15:0] @ bit48
    g1[2] = 128 << 16;        // tensor_dim1[15:0] @ bit80
    g1[3] = 128 << 16;        // tile_dim0 @ bit112
    g1[4] = 128;              // tile_dim1 @ bit128
    g1[5] = 128;              // tensor_dim0_stride[31:0] @ bit160
    g1[6] = 0;
    g1[7] = 0;
    tdm_i4 gz = {0, 0, 0, 0};
#if __clang_major__ >= 23
    tdm_i8 gz8 = {0, 0, 0, 0, 0, 0, 0, 0};
    __builtin_amdgcn_tensor_load_to_lds(g0, g1, gz, gz, gz8, 0);
#else
    __builtin_amdgcn_tensor_load_to_lds(g0, g1, gz, gz, 0);
#endif
}
#endif

__global__ __launch_bounds__(256) void gat_main(
    const float* __restrict__ h0, const float* __restrict__ adj,
    const float* __restrict__ att_b, const float* __restrict__ ff_b1,
    const float* __restrict__ ff_b2, const float* __restrict__ rezero,
    const char* __restrict__ ws, float* __restrict__ out)
{
    extern __shared__ char smem[];
    _Float16* qS  = (_Float16*)smem;                 // [128][264]; stage0: f32 h0 staging
    _Float16* kS  = qS + 128 * QS_STRIDE;            // stage0: f32 adj staging
    _Float16* vS  = kS + 128 * QS_STRIDE;
    _Float16* h0S = vS + 128 * QS_STRIDE;            // [128][136] (stage 1 only)
    _Float16* aS  = h0S;                             // aliased: attn matrix (stage 2)
    _Float16* sS  = h0S + 128 * AS_STRIDE;           // [128][40] score0 f16
    _Float16* sF  = sS + 128 * SS_STRIDE;            // [128][40] ff intermediate
    float*    cSum = (float*)(sF + 128 * SS_STRIDE); // [128]
    float*    Pv   = cSum + 128;                     // P[128] R[128] LNB[128]
    unsigned char* maskS = (unsigned char*)(Pv + 384); // [128][128] adjacency mask

    const _Float16* wsF    = (const _Float16*)ws;
    const _Float16* wsAttS = wsF + 98304;
    const _Float16* wsAttV = wsF + 102400;
    const _Float16* wsFF1  = wsF + 103424;
    const _Float16* wsFF2  = wsF + 104448;
    const float*    wsG    = (const float*)(ws + 210944);

    const int bsl = blockIdx.x;
    const int t   = threadIdx.x;
    const int wv  = t >> 5;          // wave 0..7
    const int ln  = t & 31;
    const int lm  = ln & 15;         // lane in 16-group
    const int lg  = ln >> 4;         // group 0/1

    const float Sww = wsG[384], Swb = wsG[385], Sbb = wsG[386];
    const float* h0p  = h0  + (size_t)bsl * NN * NN;
    const float* adjp = adj + (size_t)bsl * NN * NN;

    // ---- Stage 0a: async-stage h0 + adj tiles into LDS via TDM ----
#ifdef HAVE_TDM
    if (wv == 0) {
        tdm_load_128x128_f32(h0p, 0u);
        tdm_load_128x128_f32(adjp, (unsigned)(128 * QS_STRIDE * 2));
        __builtin_amdgcn_s_wait_tensorcnt(0);
    }
#else
    for (int idx = t; idx < (NN * NN) / 4; idx += 256) {
        ((float4*)qS)[idx] = ((const float4*)h0p)[idx];
        ((float4*)kS)[idx] = ((const float4*)adjp)[idx];
    }
#endif
    for (int idx = t; idx < 384; idx += 256) Pv[idx] = wsG[idx];
    __syncthreads();

    // ---- Stage 0b: staging f32 -> h0 f16 + mask bytes ----
    {
        const float* stgH = (const float*)qS;
        const float* stgA = (const float*)kS;
        for (int idx = t; idx < (NN * NN) / 4; idx += 256) {
            float4 x = ((const float4*)stgH)[idx];
            float4 a4 = ((const float4*)stgA)[idx];
            int base = idx * 4, row = base >> 7, col = base & 127;
            _Float16* d = h0S + row * AS_STRIDE + col;
            d[0] = (_Float16)x.x; d[1] = (_Float16)x.y;
            d[2] = (_Float16)x.z; d[3] = (_Float16)x.w;
            unsigned m0 = ((a4.x - ((row == col + 0) ? 1.f : 0.f)) < 0.1f) ? 1u : 0u;
            unsigned m1 = ((a4.y - ((row == col + 1) ? 1.f : 0.f)) < 0.1f) ? 1u : 0u;
            unsigned m2 = ((a4.z - ((row == col + 2) ? 1.f : 0.f)) < 0.1f) ? 1u : 0u;
            unsigned m3 = ((a4.w - ((row == col + 3) ? 1.f : 0.f)) < 0.1f) ? 1u : 0u;
            ((unsigned*)maskS)[idx] = m0 | (m1 << 8) | (m2 << 16) | (m3 << 24);
        }
    }
    __syncthreads();

    // ---- Stage 1: q/k/v = h0 @ {Wl,-Wr,Wv}^T  (384 16x16 tiles, K=128) ----
    for (int tile = wv; tile < 384; tile += 8) {
        int outm = tile >> 7;            // 0=q 1=k 2=v
        int rem  = tile & 127;
        int mt = rem >> 4, nt = rem & 15;
        const _Float16* Wf = wsF + outm * 32768;
        v8f acc = {0.f, 0.f, 0.f, 0.f, 0.f, 0.f, 0.f, 0.f};
        int arow = (mt * 16 + lm) * AS_STRIDE;
        const _Float16* brow = Wf + (nt * 16 + lm) * 128;
        __builtin_prefetch(brow + 8 * 128, 0, 1);   // next wave-stride B rows
        for (int kk = 0; kk < 128; kk += 32) {
            Frag a, b;
            int ka = kk + lg * 8;
            a.h[0] = *(const v8h*)(h0S + arow + ka);
            a.h[1] = *(const v8h*)(h0S + arow + ka + 16);
            int kb = kk + lg * 8;
            b.h[0] = *(const v8h*)(brow + kb);
            b.h[1] = *(const v8h*)(brow + kb + 16);
            acc = wmma_f16(a, b, acc);
        }
        _Float16* dst = (outm == 0) ? qS : (outm == 1) ? kS : vS;
        int col = nt * 16 + lm, rb = mt * 16 + lg * 8;
        for (int r = 0; r < 8; r++)
            dst[(rb + r) * QS_STRIDE + col] = (_Float16)acc[r];
    }
    __syncthreads();

    // ---- Stage 2: per head ----
    for (int h = 0; h < NH; h++) {
        if (t < 128) cSum[t] = 0.f;
        __syncthreads();

        // 2a: score = q_h k_h^T ; mask/exp/rowsum -> cSum
        {
            int n0 = wv * 16;
            for (int mt = 0; mt < 8; mt++) {
                int m0 = mt * 16;
                v8f sc = {0.f, 0.f, 0.f, 0.f, 0.f, 0.f, 0.f, 0.f};
                Frag a, b;
                int arow = (m0 + lm) * QS_STRIDE + h * FH_;
                a.h[0] = *(const v8h*)(qS + arow + lg * 8);
                a.h[1] = *(const v8h*)(qS + arow + lg * 8 + 16);
                int brow = (n0 + lm) * QS_STRIDE + h * FH_ + lg * 8;
                b.h[0] = *(const v8h*)(kS + brow);
                b.h[1] = *(const v8h*)(kS + brow + 16);
                sc = wmma_f16(a, b, sc);
                int row0 = m0 + lg * 8, col = n0 + lm;
                for (int r = 0; r < 8; r++) {
                    int i = row0 + r;
                    float e = maskS[i * NN + col] ? 0.f
                                                  : __expf(sc[r] * INV_SCALE);
                    e += __shfl_xor(e, 1);
                    e += __shfl_xor(e, 2);
                    e += __shfl_xor(e, 4);
                    e += __shfl_xor(e, 8);
                    if (lm == 0) atomicAdd(&cSum[i], e);
                }
            }
        }
        __syncthreads();

        // 2c: closed-form LayerNorm + softmax -> aS (f16)
        {
            int row = t >> 1, half = t & 1, j0 = half * 64;
            float c  = cSum[row];
            float is = rsqrtf(c * c * Sww + 2.f * c * Swb + Sbb + 1e-5f);
            float zmax = -1e30f;
            for (int j = j0; j < j0 + 64; j++)
                zmax = fmaxf(zmax, (c * Pv[j] + Pv[128 + j]) * is + Pv[256 + j]);
            zmax = fmaxf(zmax, __shfl_xor(zmax, 1));
            float se = 0.f;
            for (int j = j0; j < j0 + 64; j++)
                se += __expf((c * Pv[j] + Pv[128 + j]) * is + Pv[256 + j] - zmax);
            se += __shfl_xor(se, 1);
            float inv = 1.f / se;
            for (int j = j0; j < j0 + 64; j++) {
                float z = (c * Pv[j] + Pv[128 + j]) * is + Pv[256 + j];
                aS[row * AS_STRIDE + j] = (_Float16)(__expf(z - zmax) * inv);
            }
        }
        __syncthreads();

        // 2d: score0 = a @ attS^T + v_h @ attV^T + att_b  (wave w owns rows 16w..16w+15)
        int m0 = wv * 16;
        v8f s0[2];
        for (int nf = 0; nf < 2; nf++) {
            v8f acc = {0.f, 0.f, 0.f, 0.f, 0.f, 0.f, 0.f, 0.f};
            int arow = (m0 + lm) * AS_STRIDE;
            const _Float16* br = wsAttS + (nf * 16 + lm) * 128;
            for (int kk = 0; kk < 128; kk += 32) {
                Frag a, b;
                int ka = kk + lg * 8;
                a.h[0] = *(const v8h*)(aS + arow + ka);
                a.h[1] = *(const v8h*)(aS + arow + ka + 16);
                int kb = kk + lg * 8;
                b.h[0] = *(const v8h*)(br + kb);
                b.h[1] = *(const v8h*)(br + kb + 16);
                acc = wmma_f16(a, b, acc);
            }
            {   // v-part, K=32
                Frag a, b;
                int vrow = (m0 + lm) * QS_STRIDE + h * FH_;
                a.h[0] = *(const v8h*)(vS + vrow + lg * 8);
                a.h[1] = *(const v8h*)(vS + vrow + lg * 8 + 16);
                const _Float16* bv = wsAttV + (nf * 16 + lm) * 32;
                b.h[0] = *(const v8h*)(bv + lg * 8);
                b.h[1] = *(const v8h*)(bv + lg * 8 + 16);
                acc = wmma_f16(a, b, acc);
            }
            float ab = att_b[nf * 16 + lm];
            int col = nf * 16 + lm, rb = m0 + lg * 8;
            for (int r = 0; r < 8; r++) {
                acc[r] += ab;
                sS[(rb + r) * SS_STRIDE + col] = (_Float16)acc[r];
            }
            s0[nf] = acc;
        }
        __syncthreads();

        // 2e: ff1 = leaky(score0 @ ff_w1^T + b1)
        for (int nf = 0; nf < 2; nf++) {
            v8f acc = {0.f, 0.f, 0.f, 0.f, 0.f, 0.f, 0.f, 0.f};
            Frag a, b;
            int arow = (m0 + lm) * SS_STRIDE;
            a.h[0] = *(const v8h*)(sS + arow + lg * 8);
            a.h[1] = *(const v8h*)(sS + arow + lg * 8 + 16);
            const _Float16* br = wsFF1 + (nf * 16 + lm) * 32;
            b.h[0] = *(const v8h*)(br + lg * 8);
            b.h[1] = *(const v8h*)(br + lg * 8 + 16);
            acc = wmma_f16(a, b, acc);
            float bb = ff_b1[nf * 16 + lm];
            int col = nf * 16 + lm, rb = m0 + lg * 8;
            for (int r = 0; r < 8; r++) {
                float x = acc[r] + bb;
                x = fmaxf(x, 0.01f * x);
                sF[(rb + r) * SS_STRIDE + col] = (_Float16)x;
            }
        }
        __syncthreads();

        // 2f: ff2 + rezero residual -> out
        for (int nf = 0; nf < 2; nf++) {
            v8f acc = {0.f, 0.f, 0.f, 0.f, 0.f, 0.f, 0.f, 0.f};
            Frag a, b;
            int arow = (m0 + lm) * SS_STRIDE;
            a.h[0] = *(const v8h*)(sF + arow + lg * 8);
            a.h[1] = *(const v8h*)(sF + arow + lg * 8 + 16);
            const _Float16* br = wsFF2 + (nf * 16 + lm) * 32;
            b.h[0] = *(const v8h*)(br + lg * 8);
            b.h[1] = *(const v8h*)(br + lg * 8 + 16);
            acc = wmma_f16(a, b, acc);
            float b2 = ff_b2[nf * 16 + lm];
            float rz = rezero[nf * 16 + lm];
            int col = nf * 16 + lm, rb = m0 + lg * 8;
            for (int r = 0; r < 8; r++) {
                float o = s0[nf][r] + rz * (acc[r] + b2);
                out[((size_t)bsl * NN + (rb + r)) * 256 + h * FH_ + col] = o;
            }
        }
        __syncthreads();   // aS / cSum reused next head
    }
}

extern "C" void kernel_launch(void* const* d_in, const int* in_sizes, int n_in,
                              void* d_out, int out_size, void* d_ws, size_t ws_size,
                              hipStream_t stream) {
    const float* h0    = (const float*)d_in[0];
    const float* adj   = (const float*)d_in[1];
    const float* Wl    = (const float*)d_in[2];
    const float* Wr    = (const float*)d_in[3];
    const float* Wv    = (const float*)d_in[4];
    const float* sa_w  = (const float*)d_in[5];
    const float* sa_b  = (const float*)d_in[6];
    const float* ln_w  = (const float*)d_in[7];
    const float* ln_b  = (const float*)d_in[8];
    const float* att_w = (const float*)d_in[9];
    const float* att_b = (const float*)d_in[10];
    const float* ff_w1 = (const float*)d_in[11];
    const float* ff_b1 = (const float*)d_in[12];
    const float* ff_w2 = (const float*)d_in[13];
    const float* ff_b2 = (const float*)d_in[14];
    const float* rz    = (const float*)d_in[15];
    float* out = (float*)d_out;
    char*  ws  = (char*)d_ws;

    (void)hipFuncSetAttribute((const void*)gat_main,
                              hipFuncAttributeMaxDynamicSharedMemorySize, SMEM_BYTES);

    gat_prep<<<1, 256, 0, stream>>>(Wl, Wr, Wv, sa_w, sa_b, ln_w, ln_b,
                                    att_w, ff_w1, ff_w2, ws);
    gat_main<<<NB, 256, SMEM_BYTES, stream>>>(h0, adj, att_b, ff_b1, ff_b2, rz,
                                              ws, out);
}
